// AdditiveAttention_85899345920688
// MI455X (gfx1250) — matrix operand order repaired and checked by reading, not compiled, not run on hardware
//
#include <hip/hip_runtime.h>
#include <hip/hip_bf16.h>
#include <stdint.h>

// ---------------------------------------------------------------------------
// Additive attention, fused, CDNA5 (gfx1250, wave32, WMMA bf16 split-fp32,
// TDM async W1 staging with hardware LDS padding).
//   B=32, S=2048, D=1024, H=1024
// ---------------------------------------------------------------------------

#define B_LEN 32
#define S_LEN 2048
#define D_LEN 1024
#define H_LEN 1024
#define KSLABS 32          // D / 32
#define NCHUNK 64          // H / 16
#define BPAD 20            // padded u32 row stride in LDS (bank-conflict free)

typedef __attribute__((ext_vector_type(16))) __bf16 v16bf;
typedef __attribute__((ext_vector_type(8)))  float  v8f;
typedef __attribute__((ext_vector_type(4)))  unsigned int u32x4;
typedef __attribute__((ext_vector_type(8)))  int          i32x8;
typedef __attribute__((ext_vector_type(4)))  int          i32x4;

#if __has_builtin(__builtin_amdgcn_tensor_load_to_lds) && \
    __has_builtin(__builtin_amdgcn_s_wait_tensorcnt)
#define HAVE_TDM 1
#else
#define HAVE_TDM 0
#endif

union Frag {
    v16bf    v;
    uint32_t u[8];
    uint4    q[2];
};

// fp32 -> bf16 (round-to-nearest-even), returns 16-bit pattern
__device__ __forceinline__ uint32_t f32_to_bf16_rne(float x) {
    uint32_t u = __builtin_bit_cast(uint32_t, x);
    uint32_t r = u + 0x7FFFu + ((u >> 16) & 1u);
    return r >> 16;
}
__device__ __forceinline__ float bf16_bits_to_f32(uint32_t h) {
    return __builtin_bit_cast(float, h << 16);
}
// split two fp32 into packed bf16 hi-pair / lo-pair (residual) words
__device__ __forceinline__ void split_pack2(float x0, float x1,
                                            uint32_t& hi, uint32_t& lo) {
    uint32_t h0 = f32_to_bf16_rne(x0);
    uint32_t h1 = f32_to_bf16_rne(x1);
    float r0 = x0 - bf16_bits_to_f32(h0);
    float r1 = x1 - bf16_bits_to_f32(h1);
    uint32_t l0 = f32_to_bf16_rne(r0);
    uint32_t l1 = f32_to_bf16_rne(r1);
    hi = h0 | (h1 << 16);
    lo = l0 | (l1 << 16);
}

#if HAVE_TDM
// ---------------------------------------------------------------------------
// TDM: DMA one 64KB W1 slab (16384 dwords) global -> LDS, inserting 4 pad
// dwords after every 16 data dwords (hardware-applied bank-conflict padding,
// producing exactly the BPAD=20 layout the WMMA B-fragment loads expect).
// D# per CDNA5 ISA ch.8. Issued by one wave; tracked via TENSORcnt.
// ---------------------------------------------------------------------------
__device__ __forceinline__ void tdm_load_slab(const uint32_t* src,
                                              uint32_t lds_byte_off) {
    uint64_t ga = (uint64_t)(uintptr_t)src;
    u32x4 g0;
    g0[0] = 1u;                                  // count=1, user mode
    g0[1] = lds_byte_off;                        // lds_addr
    g0[2] = (uint32_t)ga;                        // global_addr[31:0]
    g0[3] = (uint32_t)(ga >> 32) | (2u << 30);   // global_addr[56:32] | type=2
    i32x8 g1;
    // data_size=2 (4B) | pad_enable | pad_interval=3 (16 dw) | pad_amount=3 (4 dw)
    g1[0] = (int)((2u << 16) | (1u << 20) | (3u << 22) | (3u << 25));
    g1[1] = (int)(16384u << 16);                 // tensor_dim0[15:0] in [31:16]
    g1[2] = (int)(1u << 16);                     // tensor_dim0 hi=0 | tensor_dim1=1
    g1[3] = (int)(16384u << 16);                 // tile_dim0 = 16384
    g1[4] = 1;                                   // tile_dim1 = 1
    g1[5] = 16384;                               // tensor_dim0_stride lo
    g1[6] = 0;
    g1[7] = 0;
    i32x4 gz = {0, 0, 0, 0};
#if defined(__clang_major__) && __clang_major__ >= 23
    i32x8 gz8 = {0, 0, 0, 0, 0, 0, 0, 0};
    __builtin_amdgcn_tensor_load_to_lds(g0, g1, gz, gz, gz8, 0);
#else
    __builtin_amdgcn_tensor_load_to_lds(g0, g1, gz, gz, 0);
#endif
}
#endif // HAVE_TDM

// ---------------------------------------------------------------------------
// 1) Pack W1 [D,H] fp32 -> bf16 hi/lo, pre-swizzled as [slab t][n][kpair p]
// ---------------------------------------------------------------------------
__global__ __launch_bounds__(256) void k_packw1(const float* __restrict__ W1,
                                                uint32_t* __restrict__ w1hi,
                                                uint32_t* __restrict__ w1lo) {
    int idx = blockIdx.x * 256 + threadIdx.x;       // 0 .. 524287
    int kp  = idx >> 10;                            // K pair index, 0..511
    int n   = idx & (H_LEN - 1);
    float x0 = W1[(size_t)(2 * kp)     * H_LEN + n];
    float x1 = W1[(size_t)(2 * kp + 1) * H_LEN + n];
    uint32_t h, l;
    split_pack2(x0, x1, h, l);
    int t = kp >> 4, p = kp & 15;                   // slab, pair-in-slab
    size_t o = ((size_t)t * H_LEN + n) * 16 + p;
    w1hi[o] = h;
    w1lo[o] = l;
}

// ---------------------------------------------------------------------------
// 2) query_proj = query @ W2   [B,H]  (tiny, fp32 FMA)
// ---------------------------------------------------------------------------
__global__ __launch_bounds__(256) void k_qproj(const float* __restrict__ query,
                                               const float* __restrict__ W2,
                                               float* __restrict__ qproj) {
    int b = blockIdx.y;
    int h = blockIdx.x * 256 + threadIdx.x;
    float s = 0.f;
    for (int d = 0; d < D_LEN; ++d)
        s = fmaf(query[b * D_LEN + d], W2[(size_t)d * H_LEN + h], s);
    qproj[b * H_LEN + h] = s;
}

// ---------------------------------------------------------------------------
// 3) Hot kernel: per block = (b, 32 s-rows).
//    keys_proj tile via split-bf16 WMMA, fused tanh(+qproj) and dot(v),
//    emitting scores[b, s0..s0+31]. Never materializes [B,S,H].
// ---------------------------------------------------------------------------
__global__ __launch_bounds__(256) void k_energy(
    const float*    __restrict__ values,  // [B,S,D]
    const uint32_t* __restrict__ w1hi,    // [KSLABS][H][16]
    const uint32_t* __restrict__ w1lo,
    const float*    __restrict__ qproj,   // [B,H]
    const float*    __restrict__ vvec,    // [H]
    float*          __restrict__ scores)  // [B,S]
{
    extern __shared__ char smem[];
    uint32_t* Bhi = (uint32_t*)(smem);                    // H*BPAD u32 = 80KB
    uint32_t* Blo = (uint32_t*)(smem + 81920);            // 80KB
    uint32_t* Ahi = (uint32_t*)(smem + 163840);           // 32*BPAD u32
    uint32_t* Alo = (uint32_t*)(smem + 166400);
    float*    sbuf = (float*)  (smem + 168960);           // 32 floats

    const int tid  = threadIdx.x;
    const int lane = tid & 31;
    const int wv   = tid >> 5;         // wave 0..7
    const int rg   = wv & 1;           // 16-row group
    const int cset = wv >> 1;          // 0..3 chunk set
    const int hf   = lane >> 4;        // lane half (0/1)
    const int ln   = lane & 15;

    const int b  = blockIdx.y;
    const int s0 = blockIdx.x * 32;

    if (tid < 32) sbuf[tid] = 0.0f;

    v8f acc[16];
#pragma unroll
    for (int i = 0; i < 16; ++i)
        acc[i] = v8f{0.f, 0.f, 0.f, 0.f, 0.f, 0.f, 0.f, 0.f};

    const float* vbase = values + ((size_t)b * S_LEN + s0) * D_LEN;

    for (int t = 0; t < KSLABS; ++t) {
        // ---- stage W1 slab into LDS (TDM DMA with hardware padding) -------
#if HAVE_TDM
        if (wv == 0) {
            tdm_load_slab(w1hi + (size_t)t * (H_LEN * 16), 0u);
            tdm_load_slab(w1lo + (size_t)t * (H_LEN * 16), 81920u);
        }
#else
        {
            const uint4* srcH = (const uint4*)(w1hi + (size_t)t * (H_LEN * 16));
            const uint4* srcL = (const uint4*)(w1lo + (size_t)t * (H_LEN * 16));
            uint4* dH = (uint4*)Bhi;
            uint4* dL = (uint4*)Blo;
#pragma unroll
            for (int j = 0; j < 16; ++j) {
                int e  = tid + 256 * j;
                int di = (e >> 2) * (BPAD / 4) + (e & 3);
                dH[di] = srcH[e];
                dL[di] = srcL[e];
            }
        }
#endif
        // prefetch next W1 slab toward L2 while we compute
        if (t + 1 < KSLABS)
            __builtin_prefetch((const void*)(w1hi + (size_t)(t + 1) * (H_LEN * 16)
                                             + (size_t)tid * 64), 0, 0);
        // ---- stage A tile: split values fp32 -> bf16 hi/lo pairs ----------
        // (runs while the TDM DMA streams the W1 slab)
        {
            int m = tid >> 3, q = tid & 7;     // 32 rows x 8 float4
            const float4 v4 = *(const float4*)(vbase + (size_t)m * D_LEN
                                               + t * 32 + 4 * q);
            uint32_t h0, l0, h1, l1;
            split_pack2(v4.x, v4.y, h0, l0);
            split_pack2(v4.z, v4.w, h1, l1);
            Ahi[m * BPAD + 2 * q]     = h0;
            Ahi[m * BPAD + 2 * q + 1] = h1;
            Alo[m * BPAD + 2 * q]     = l0;
            Alo[m * BPAD + 2 * q + 1] = l1;
        }
#if HAVE_TDM
        if (wv == 0) __builtin_amdgcn_s_wait_tensorcnt(0);
#endif
        __syncthreads();

        // ---- A fragments (16x32 bf16: lane m=ln, pairs 8hf..8hf+7)
        int m = rg * 16 + ln;
        Frag aH, aL;
        const uint4* AHq = (const uint4*)Ahi;
        const uint4* ALq = (const uint4*)Alo;
        aH.q[0] = AHq[m * (BPAD / 4) + 2 * hf];
        aH.q[1] = AHq[m * (BPAD / 4) + 2 * hf + 1];
        aL.q[0] = ALq[m * (BPAD / 4) + 2 * hf];
        aL.q[1] = ALq[m * (BPAD / 4) + 2 * hf + 1];

        const uint4* BHq = (const uint4*)Bhi;
        const uint4* BLq = (const uint4*)Blo;
        // two chunks per iteration: consecutive WMMAs hit independent
        // accumulators, hiding the bf16 WMMA->WMMA RAW hazard spacing
#pragma unroll
        for (int i = 0; i < 16; i += 2) {
            int n0 = (cset + 4 * i) * 16 + ln;
            int n1 = (cset + 4 * (i + 1)) * 16 + ln;
            Frag bH0, bL0, bH1, bL1;
            int bi0 = n0 * (BPAD / 4) + 2 * hf;
            int bi1 = n1 * (BPAD / 4) + 2 * hf;
            bH0.q[0] = BHq[bi0];     bH0.q[1] = BHq[bi0 + 1];
            bL0.q[0] = BLq[bi0];     bL0.q[1] = BLq[bi0 + 1];
            bH1.q[0] = BHq[bi1];     bH1.q[1] = BHq[bi1 + 1];
            bL1.q[0] = BLq[bi1];     bL1.q[1] = BLq[bi1 + 1];
            // C += Ahi*Bhi + Ahi*Blo + Alo*Bhi   (split-fp32 emulation)
            acc[i]     = __builtin_amdgcn_wmma_f32_16x16x32_bf16(
                false, aH.v, false, bH0.v, (short)0, acc[i],     false, false);
            acc[i + 1] = __builtin_amdgcn_wmma_f32_16x16x32_bf16(
                false, aH.v, false, bH1.v, (short)0, acc[i + 1], false, false);
            acc[i]     = __builtin_amdgcn_wmma_f32_16x16x32_bf16(
                false, aH.v, false, bL0.v, (short)0, acc[i],     false, false);
            acc[i + 1] = __builtin_amdgcn_wmma_f32_16x16x32_bf16(
                false, aH.v, false, bL1.v, (short)0, acc[i + 1], false, false);
            acc[i]     = __builtin_amdgcn_wmma_f32_16x16x32_bf16(
                false, aL.v, false, bH0.v, (short)0, acc[i],     false, false);
            acc[i + 1] = __builtin_amdgcn_wmma_f32_16x16x32_bf16(
                false, aL.v, false, bH1.v, (short)0, acc[i + 1], false, false);
        }
        __syncthreads();
    }

    // ---- fused epilogue: score[s] += v[n] * tanh(C + qproj[n]) ------------
    float part[8];
#pragma unroll
    for (int r = 0; r < 8; ++r) part[r] = 0.f;
#pragma unroll
    for (int i = 0; i < 16; ++i) {
        int n = (cset + 4 * i) * 16 + ln;
        float qp = qproj[b * H_LEN + n];
        float vn = vvec[n];
#pragma unroll
        for (int r = 0; r < 8; ++r)
            part[r] += vn * tanhf(acc[i][r] + qp);
    }
    // reduce over the 16 lanes of each half (N dimension)
#pragma unroll
    for (int r = 0; r < 8; ++r) {
        part[r] += __shfl_xor(part[r], 1, 32);
        part[r] += __shfl_xor(part[r], 2, 32);
        part[r] += __shfl_xor(part[r], 4, 32);
        part[r] += __shfl_xor(part[r], 8, 32);
    }
    if (ln == 0) {
#pragma unroll
        for (int r = 0; r < 8; ++r)
            atomicAdd(&sbuf[rg * 16 + 8 * hf + r], part[r]);  // ds_add_f32
    }
    __syncthreads();
    if (tid < 32)
        scores[(size_t)b * S_LEN + s0 + tid] = sbuf[tid];
}

// ---------------------------------------------------------------------------
// 4) softmax over S per batch row -> attention weights (written to d_out)
// ---------------------------------------------------------------------------
__global__ __launch_bounds__(256) void k_softmax(const float* __restrict__ scores,
                                                 float* __restrict__ weights) {
    __shared__ float red[256];
    int b = blockIdx.x, tid = threadIdx.x;
    const float* row = scores + (size_t)b * S_LEN;
    float mx = -INFINITY;
    for (int s = tid; s < S_LEN; s += 256) mx = fmaxf(mx, row[s]);
    red[tid] = mx; __syncthreads();
    for (int o = 128; o > 0; o >>= 1) {
        if (tid < o) red[tid] = fmaxf(red[tid], red[tid + o]);
        __syncthreads();
    }
    mx = red[0]; __syncthreads();
    float sm = 0.f;
    for (int s = tid; s < S_LEN; s += 256) sm += expf(row[s] - mx);
    red[tid] = sm; __syncthreads();
    for (int o = 128; o > 0; o >>= 1) {
        if (tid < o) red[tid] += red[tid + o];
        __syncthreads();
    }
    float inv = 1.0f / red[0];
    for (int s = tid; s < S_LEN; s += 256)
        weights[(size_t)b * S_LEN + s] = expf(row[s] - mx) * inv;
}

// ---------------------------------------------------------------------------
// 5) context partials: ws[sc][b][d] = sum_{s in chunk sc} w[b,s] * values[b,s,d]
// ---------------------------------------------------------------------------
__global__ __launch_bounds__(256) void k_ctx(const float* __restrict__ values,
                                             const float* __restrict__ weights,
                                             float* __restrict__ partial) {
    int sc = blockIdx.x, b = blockIdx.y, tid = threadIdx.x;
    float accd[4] = {0.f, 0.f, 0.f, 0.f};
    for (int s = sc * 128; s < sc * 128 + 128; ++s) {
        float w = weights[(size_t)b * S_LEN + s];
        const float* vr = values + ((size_t)b * S_LEN + s) * D_LEN;
#pragma unroll
        for (int j = 0; j < 4; ++j)
            accd[j] = fmaf(w, vr[tid + 256 * j], accd[j]);
    }
    float* out = partial + ((size_t)sc * B_LEN + b) * D_LEN;
#pragma unroll
    for (int j = 0; j < 4; ++j) out[tid + 256 * j] = accd[j];
}

// 6) reduce the 16 s-chunks -> context [B,D]
__global__ __launch_bounds__(256) void k_ctxred(const float* __restrict__ partial,
                                                float* __restrict__ ctx) {
    int i = blockIdx.x * 256 + threadIdx.x;   // b*D + d, 0..32767
    float s = 0.f;
    for (int sc = 0; sc < 16; ++sc)
        s += partial[(size_t)sc * (B_LEN * D_LEN) + i];
    ctx[i] = s;
}

// ---------------------------------------------------------------------------
extern "C" void kernel_launch(void* const* d_in, const int* in_sizes, int n_in,
                              void* d_out, int out_size, void* d_ws, size_t ws_size,
                              hipStream_t stream) {
    const float* query  = (const float*)d_in[0];   // [B,D]
    const float* values = (const float*)d_in[1];   // [B,S,D]
    const float* W1     = (const float*)d_in[2];   // [D,H]
    const float* W2     = (const float*)d_in[3];   // [D,H]
    const float* vvec   = (const float*)d_in[4];   // [H]

    float* ctx_out = (float*)d_out;                         // [B,D]   = 32768
    float* att_out = (float*)d_out + B_LEN * D_LEN;         // [B,S]   = 65536

    char* ws = (char*)d_ws;
    float*    qproj   = (float*)(ws);                       // 128 KB
    float*    scores  = (float*)(ws + 131072);              // 256 KB
    uint32_t* w1hi    = (uint32_t*)(ws + 393216);           // 2 MB
    uint32_t* w1lo    = (uint32_t*)(ws + 2490368);          // 2 MB
    float*    ctxpart = (float*)(ws + 4587520);             // 2 MB

    k_packw1<<<dim3(2048), dim3(256), 0, stream>>>(W1, w1hi, w1lo);
    k_qproj<<<dim3(H_LEN / 256, B_LEN), dim3(256), 0, stream>>>(query, W2, qproj);

    const size_t smem_bytes = 169088;  // Bhi/Blo 80KB each + A tiles + sbuf
    k_energy<<<dim3(S_LEN / 32, B_LEN), dim3(256), smem_bytes, stream>>>(
        values, w1hi, w1lo, qproj, vvec, scores);

    k_softmax<<<dim3(B_LEN), dim3(256), 0, stream>>>(scores, att_out);
    k_ctx<<<dim3(16, B_LEN), dim3(256), 0, stream>>>(values, att_out, ctxpart);
    k_ctxred<<<dim3((B_LEN * D_LEN) / 256), dim3(256), 0, stream>>>(ctxpart, ctx_out);
}
